// ConvP4_12987981103166
// MI455X (gfx1250) — matrix operand order repaired and checked by reading, not compiled, not run on hardware
//
#include <hip/hip_runtime.h>
#include <hip/hip_bf16.h>

// ---------------------------------------------------------------------------
// P4 group-equivariant conv as implicit GEMM on CDNA5 WMMA (bf16 -> f32 acc).
// Activation rows staged in LDS via global_load_async_to_lds_b128 (+
// s_wait_asynccnt), next group-plane prefetched via global_prefetch_b8.
//   x:      [8][4][128][128][64]  f32
//   kernel: [3][3][3][64][128]    f32
//   out:    [8][4][128][128][128] f32
// ---------------------------------------------------------------------------

typedef __attribute__((ext_vector_type(16))) __bf16        v16bf;
typedef __attribute__((ext_vector_type(8)))  float         v8f;
typedef __attribute__((ext_vector_type(8)))  unsigned int  v8u;
typedef __attribute__((ext_vector_type(4)))  unsigned int  v4u;
typedef __attribute__((ext_vector_type(4)))  int           v4i;

#define BDIM 8
#define GDIM 4
#define HDIM 128
#define WDIM 128
#define CIN  64
#define FDIM 128

// LDS pixel stride: 64 data ushorts padded to 72 (144B = 36 dwords) so that
// lanes 0..15 (stride 36 dwords) land on distinct bank groups, 16B-aligned.
#define LPIX 72
#define LROW (WDIM * LPIX)            // 9216 ushorts per staged row

#define XB_ELEMS ((size_t)BDIM * GDIM * HDIM * WDIM * CIN)   // 33,554,432
#define WT_ELEMS ((size_t)4 * 27 * 2 * FDIM * 32)            //    884,736

#if __has_builtin(__builtin_amdgcn_global_load_async_to_lds_b128)
typedef __attribute__((address_space(1))) v4i* as1_v4i_p;    // global int4*
typedef __attribute__((address_space(3))) v4i* as3_v4i_p;    // LDS int4*
#define HAVE_ASYNC_LDS 1
#endif

__device__ __forceinline__ unsigned short f2bf(float f) {
  unsigned u = __float_as_uint(f);
  unsigned r = u + 0x7FFFu + ((u >> 16) & 1u);   // round-to-nearest-even
  return (unsigned short)(r >> 16);
}

// ---- pass 1: x f32 -> bf16 -------------------------------------------------
__global__ void cvt_x_kernel(const float* __restrict__ x,
                             unsigned short* __restrict__ xb, int n4) {
  int i = blockIdx.x * blockDim.x + threadIdx.x;
  if (i >= n4) return;
  float4 v = ((const float4*)x)[i];
  ushort4 o;
  o.x = f2bf(v.x); o.y = f2bf(v.y); o.z = f2bf(v.z); o.w = f2bf(v.w);
  ((ushort4*)xb)[i] = o;
}

// ---- pass 2: build rotated, B-fragment-friendly weights --------------------
// wt[j][tap=k*9+kh*3+kw][cc][f][ci] = rot90(kernel,j)[kh,kw,k, cc*32+ci, f]
__global__ void prep_w_kernel(const float* __restrict__ ker,
                              unsigned short* __restrict__ wt) {
  int idx = blockIdx.x * blockDim.x + threadIdx.x;
  if (idx >= (int)WT_ELEMS) return;
  int ci = idx & 31;
  int t  = idx >> 5;
  int f  = t & 127; t >>= 7;
  int cc = t & 1;   t >>= 1;
  int tap = t % 27;
  int j   = t / 27;
  int k  = tap / 9;
  int r  = tap % 9;
  int kh = r / 3, kw = r % 3;
  int skh, skw;
  switch (j) {                       // indices of rot90(kernel, j)[kh][kw]
    case 0:  skh = kh;     skw = kw;     break;
    case 1:  skh = kw;     skw = 2 - kh; break;
    case 2:  skh = 2 - kh; skw = 2 - kw; break;
    default: skh = 2 - kw; skw = kh;     break;
  }
  int cin = cc * 32 + ci;
  float v = ker[((((skh * 3 + skw) * 3 + k) * CIN + cin) * FDIM) + f];
  wt[idx] = f2bf(v);
}

// ---- pass 3: WMMA implicit GEMM with LDS-staged activations ----------------
__global__ __launch_bounds__(256)
void p4conv_wmma_kernel(const unsigned short* __restrict__ xb,
                        const unsigned short* __restrict__ wt,
                        float* __restrict__ out) {
  __shared__ __align__(16) unsigned short ldsA[3 * LROW];   // 54 KB

  const int blk = blockIdx.x;          // 4096 = B * 4 * H
  const int h = blk & 127;
  const int j = (blk >> 7) & 3;
  const int b = blk >> 9;

  const int tid  = threadIdx.x;
  const int lane = tid & 31;
  const int wave = tid >> 5;
  const int mg = wave & 3;             // 4 M-groups of 32 pixels
  const int ng = wave >> 2;            // 2 N-groups of 64 channels
  const int lm = lane & 15;
  const int hi = (lane >> 4) & 1;      // 0: lanes 0-15, 1: lanes 16-31

  v16bf azero;
  #pragma unroll
  for (int i = 0; i < 16; ++i) azero[i] = (__bf16)0.0f;

  v8f acc[2][4];
  #pragma unroll
  for (int s = 0; s < 2; ++s)
    #pragma unroll
    for (int n = 0; n < 4; ++n) acc[s][n] = {};

  for (int k = 0; k < 3; ++k) {
    const int g = (j + k + 3) & 3;     // (j + k - 1) mod 4
    const unsigned short* xplane =
        xb + (size_t)(b * GDIM + g) * HDIM * WDIM * CIN;

    // ---- stage the 3 needed rows of this group plane into LDS ----
    __syncthreads();                   // previous k's readers done
    for (int kh = 0; kh < 3; ++kh) {
      const int sh = h + kh - 1;
      if (sh < 0 || sh >= HDIM) continue;          // tap skipped below too
      const unsigned short* src = xplane + (size_t)sh * WDIM * CIN;
      // 128 px * 8 chunks of 16B each = 1024 chunks, 256 threads * 4 iters
      for (int c = tid; c < WDIM * 8; c += 256) {
        const int pix = c >> 3;
        const int part = c & 7;
#ifdef HAVE_ASYNC_LDS
        __builtin_amdgcn_global_load_async_to_lds_b128(
            (as1_v4i_p)(src + (size_t)pix * CIN + part * 8),
            (as3_v4i_p)(&ldsA[kh * LROW + pix * LPIX + part * 8]),
            0, 0);
#else
        *(v4u*)(&ldsA[kh * LROW + pix * LPIX + part * 8]) =
            *(const v4u*)(src + (size_t)pix * CIN + part * 8);
#endif
      }
    }
#ifdef HAVE_ASYNC_LDS
#if __has_builtin(__builtin_amdgcn_s_wait_asynccnt)
    __builtin_amdgcn_s_wait_asynccnt(0);
#else
    asm volatile("s_wait_asynccnt 0x0" ::: "memory");
#endif
#endif
    __syncthreads();

    // ---- prefetch next group plane's rows (global_prefetch_b8) ----
    if (k < 2) {
      const int gn = (j + k + 4) & 3;
      const unsigned short* nplane =
          xb + (size_t)(b * GDIM + gn) * HDIM * WDIM * CIN;
      for (int kh = 0; kh < 3; ++kh) {
        const int sh = h + kh - 1;
        if (sh < 0 || sh >= HDIM) continue;
        const unsigned short* src = nplane + (size_t)sh * WDIM * CIN;
        for (int c = tid; c < 128; c += 256)       // 128 cachelines of 128B
          __builtin_prefetch(src + (size_t)c * 64, 0, 3);
      }
    }

    for (int kh = 0; kh < 3; ++kh) {
      const int sh = h + kh - 1;
      if (sh < 0 || sh >= HDIM) continue;          // whole tap contributes 0
      const unsigned short* lrow = &ldsA[kh * LROW];
      for (int kw = 0; kw < 3; ++kw) {
        const int tap = k * 9 + kh * 3 + kw;
        const unsigned short* wtap =
            wt + ((size_t)j * 27 + tap) * 2 * (FDIM * 32);
        #pragma unroll
        for (int cc = 0; cc < 2; ++cc) {
          // B fragments: 32x16 bf16, lane = column f, 16 contiguous K values
          const unsigned short* wbase = wtap + (size_t)cc * (FDIM * 32);
          v16bf bfrag[4];
          #pragma unroll
          for (int n = 0; n < 4; ++n) {
            const int f = ng * 64 + n * 16 + lm;
            const unsigned short* p = wbase + (size_t)f * 32 + hi * 16;
            bfrag[n] = __builtin_bit_cast(v16bf, *(const v8u*)p);
          }
          // A fragments from LDS: 16x32 bf16, lane holds row M=lm,
          //   K = {off..off+15}, off = hi*16  (matches B)
          v16bf afrag[2];
          #pragma unroll
          for (int s = 0; s < 2; ++s) {
            const int wpix = mg * 32 + s * 16 + lm + kw - 1;
            if (wpix >= 0 && wpix < WDIM) {
              const unsigned short* p = lrow + wpix * LPIX + cc * 32 + hi * 16;
              v4u lo = *(const v4u*)p;
              v4u hh = *(const v4u*)(p + 8);
              v8u raw;
              raw[0] = lo[0]; raw[1] = lo[1]; raw[2] = lo[2]; raw[3] = lo[3];
              raw[4] = hh[0]; raw[5] = hh[1]; raw[6] = hh[2]; raw[7] = hh[3];
              afrag[s] = __builtin_bit_cast(v16bf, raw);
            } else {
              afrag[s] = azero;                    // SAME zero padding in W
            }
          }
          // 8 x v_wmma_f32_16x16x32_bf16
          #pragma unroll
          for (int s = 0; s < 2; ++s)
            #pragma unroll
            for (int n = 0; n < 4; ++n)
              acc[s][n] = __builtin_amdgcn_wmma_f32_16x16x32_bf16(
                  false, afrag[s], false, bfrag[n],
                  (short)0, acc[s][n], false, false);
        }
      }
    }
  }

  // D layout: VGPR r -> row (r + hi*8), lane -> column lm
  float* obase = out + (((size_t)(b * GDIM + j) * HDIM + h) * WDIM) * FDIM;
  #pragma unroll
  for (int s = 0; s < 2; ++s) {
    const int w0 = mg * 32 + s * 16 + hi * 8;
    #pragma unroll
    for (int n = 0; n < 4; ++n) {
      const int f = ng * 64 + n * 16 + lm;
      #pragma unroll
      for (int r = 0; r < 8; ++r)
        obase[(size_t)(w0 + r) * FDIM + f] = acc[s][n][r];
    }
  }
}

extern "C" void kernel_launch(void* const* d_in, const int* in_sizes, int n_in,
                              void* d_out, int out_size, void* d_ws, size_t ws_size,
                              hipStream_t stream) {
  const float* x   = (const float*)d_in[0];
  const float* ker = (const float*)d_in[1];
  float* out = (float*)d_out;

  unsigned short* xb = (unsigned short*)d_ws;       // 67.1 MB bf16 x
  unsigned short* wt = xb + XB_ELEMS;               //  1.8 MB bf16 weights

  const int n4 = (int)(XB_ELEMS / 4);
  cvt_x_kernel<<<(n4 + 255) / 256, 256, 0, stream>>>(x, xb, n4);

  const int nw = (int)WT_ELEMS;
  prep_w_kernel<<<(nw + 255) / 256, 256, 0, stream>>>(ker, wt);

  p4conv_wmma_kernel<<<BDIM * GDIM * HDIM, 256, 0, stream>>>(xb, wt, out);
}
